// QIFNetwork_47115791237686
// MI455X (gfx1250) — matrix operand order repaired and checked by reading, not compiled, not run on hardware
//
#include <hip/hip_runtime.h>
#include <hip/hip_bf16.h>

// ---------------------------------------------------------------------------
// QIF network step for MI455X (gfx1250, wave32, WMMA).
//   B=8192, H=2048, I=1024, O=512
// Pipeline:
//   1) f32 -> bf16 conversions (current, w_i2h, w_h2o)
//   2) eta = current @ w_i2h^T + b_i2h + b_h2h          (WMMA bf16->f32)
//      - tiles staged to LDS with GLOBAL_LOAD_ASYNC_TO_LDS_B128 (ASYNCcnt)
//   3) fused neuron kernel: next_spike, min/argmin, update_potential,
//      w_h2h column gather, v/t/min_dt/k outputs, clipped bf16 v for GEMM2
//   4) output_potential = clip(v) @ w_h2o^T + b_h2o     (WMMA bf16->f32)
//      output_spike = w_h2o[:, k] gather
// ---------------------------------------------------------------------------

typedef __attribute__((ext_vector_type(16))) __bf16 v16bf;
typedef __attribute__((ext_vector_type(8)))  float  v8f;

union FragBF16 {
  v16bf v;
  uint4 q[2];
};

#define QIF_B 8192
#define QIF_H 2048
#define QIF_I 1024
#define QIF_O 512
#define QIF_EPS 1e-6f
#define QIF_TINF 10000.0f
#define QIF_PI 3.14159265358979323846f

// Low 32 bits of a flat shared-aperture address are the LDS byte offset
// (ISA 10.2).
__device__ __forceinline__ unsigned qif_lds_addr(const void* p) {
  return (unsigned)(size_t)p;
}

// Async global -> LDS copy of 16 bytes (GLOBAL_LOAD_ASYNC_TO_LDS_B128,
// tracked with ASYNCcnt; no VGPR data path).
__device__ __forceinline__ void qif_async_copy_b128(unsigned lds_addr,
                                                    const void* gptr) {
  asm volatile("global_load_async_to_lds_b128 %0, %1, off"
               :
               : "v"(lds_addr), "v"(gptr)
               : "memory");
}

__device__ __forceinline__ void qif_wait_asynccnt0() {
  asm volatile("s_wait_asynccnt 0x0" ::: "memory");
}

// ---------------------------------------------------------------------------
// f32 -> bf16 conversion (grid-stride)
// ---------------------------------------------------------------------------
__global__ __launch_bounds__(256) void f32_to_bf16_kernel(
    const float* __restrict__ src, __bf16* __restrict__ dst, int n) {
  int i = blockIdx.x * blockDim.x + threadIdx.x;
  int stride = gridDim.x * blockDim.x;
  for (; i < n; i += stride) dst[i] = (__bf16)src[i];
}

// ---------------------------------------------------------------------------
// Tiled WMMA GEMM:  C[M,N] = A[M,K] * Bm[N,K]^T + bias0[N] (+ bias1[N])
// Block tile 128x128, K-step 32, 8 waves (256 threads), 4x2 wave grid,
// each wave computes a 32x64 tile = 2x4 WMMA 16x16 accumulators.
// Double-buffered LDS staging via async global->LDS copies.
// ---------------------------------------------------------------------------
__global__ __launch_bounds__(256) void gemm_bf16_wmma(
    const __bf16* __restrict__ A,    // [M,K] row-major
    const __bf16* __restrict__ Bm,   // [N,K] row-major (i.e. B^T with K inner)
    const float* __restrict__ bias0, // [N] or null
    const float* __restrict__ bias1, // [N] or null
    float* __restrict__ C,           // [M,N]
    int M, int N, int K) {
  __shared__ __bf16 lsA[2][128][32];  // 16 KB
  __shared__ __bf16 lsB[2][128][32];  // 16 KB

  const int tid  = threadIdx.x;
  const int wave = tid >> 5;
  const int lane = tid & 31;
  const int wm   = wave & 3;   // wave row   (4 waves over M: 32 rows each)
  const int wn   = wave >> 2;  // wave col   (2 waves over N: 64 cols each)
  const int lrow = lane & 15;  // lane within 16-wide group
  const int hsel = lane >> 4;  // 0 = lanes 0-15, 1 = lanes 16-31
  const int m0 = blockIdx.y * 128;
  const int n0 = blockIdx.x * 128;
  const int nk = K >> 5;

  // Per-thread staging coordinates: each thread copies 16B from A-tile and
  // 16B from B-tile in each of 2 rounds => full 128x32 bf16 tiles.
  const int srow = tid >> 2;        // 0..63
  const int scol = (tid & 3) << 3;  // 0,8,16,24

  auto stage = [&](int buf, int kt) {
    const int k0 = kt << 5;
#pragma unroll
    for (int r = 0; r < 2; ++r) {
      const int rr = srow + (r << 6);  // 0..127
      qif_async_copy_b128(qif_lds_addr(&lsA[buf][rr][scol]),
                          &A[(size_t)(m0 + rr) * K + k0 + scol]);
      qif_async_copy_b128(qif_lds_addr(&lsB[buf][rr][scol]),
                          &Bm[(size_t)(n0 + rr) * K + k0 + scol]);
    }
  };

  v8f acc[2][4] = {};

  stage(0, 0);
  qif_wait_asynccnt0();
  __syncthreads();

  int cur = 0;
  for (int kt = 0; kt < nk; ++kt) {
    if (kt + 1 < nk) stage(cur ^ 1, kt + 1);

    // A fragments: 16x32 bf16.  ISA layout (dense 16-bit A):
    //   lanes 0-15 : K 0..7 (VGPR0-3), K 16..23 (VGPR4-7)
    //   lanes 16-31: K 8..15,           K 24..31
    FragBF16 afr[2];
#pragma unroll
    for (int tm = 0; tm < 2; ++tm) {
      const char* p = (const char*)&lsA[cur][wm * 32 + tm * 16 + lrow][0];
      afr[tm].q[0] = *(const uint4*)(p + hsel * 16);
      afr[tm].q[1] = *(const uint4*)(p + 32 + hsel * 16);
    }
    // B fragments: 32x16 bf16, N on lanes, K packed:
    //   lanes 0-15 hold K 0..7, 16..23; lanes 16-31 hold K 8..15, 24..31
    FragBF16 bfr[4];
#pragma unroll
    for (int tn = 0; tn < 4; ++tn) {
      const char* p = (const char*)&lsB[cur][wn * 64 + tn * 16 + lrow][0];
      bfr[tn].q[0] = *(const uint4*)(p + hsel * 16);
      bfr[tn].q[1] = *(const uint4*)(p + 32 + hsel * 16);
    }
#pragma unroll
    for (int tm = 0; tm < 2; ++tm)
#pragma unroll
      for (int tn = 0; tn < 4; ++tn)
        acc[tm][tn] = __builtin_amdgcn_wmma_f32_16x16x32_bf16(
            false, afr[tm].v, false, bfr[tn].v, (short)0, acc[tm][tn], false,
            false);

    // Make this iteration's async staging (other buffer) visible to all
    // waves before it is consumed next iteration.
    qif_wait_asynccnt0();
    __syncthreads();
    cur ^= 1;
  }

  // Epilogue: 16x16 f32 C layout: VGPR r -> M = r (lanes 0-15) or 8+r.
#pragma unroll
  for (int tm = 0; tm < 2; ++tm) {
#pragma unroll
    for (int tn = 0; tn < 4; ++tn) {
      const int mbase = m0 + wm * 32 + tm * 16 + hsel * 8;
      const int n = n0 + wn * 64 + tn * 16 + lrow;
      float bv = 0.0f;
      if (bias0) bv += bias0[n];
      if (bias1) bv += bias1[n];
#pragma unroll
      for (int r = 0; r < 8; ++r) {
        C[(size_t)(mbase + r) * N + n] = acc[tm][tn][r] + bv;
      }
    }
  }
}

// ---------------------------------------------------------------------------
// Fused neuron kernel: one block (256 threads) per batch row.
// Each thread owns 8 of the 2048 hidden units (h = tid + j*256).
// ---------------------------------------------------------------------------
__device__ __forceinline__ float qif_sign(float x) {
  return (x > 0.0f) ? 1.0f : ((x < 0.0f) ? -1.0f : 0.0f);
}

__global__ __launch_bounds__(256) void neuron_kernel(
    const float* __restrict__ spike, const float* __restrict__ v_in,
    const float* __restrict__ eta_arr, const float* __restrict__ w_h2h,
    const float* __restrict__ t_in, const float* __restrict__ syn_delay_p,
    const float* __restrict__ vmin_p, const float* __restrict__ vmax_p,
    float* __restrict__ v_out, __bf16* __restrict__ vclip_bf,
    float* __restrict__ t_out, float* __restrict__ mindt_out,
    float* __restrict__ k_out_as_f32) {
  const int b = blockIdx.x;
  const int tid = threadIdx.x;
  const float syn_delay = *syn_delay_p;
  const float vmin = *vmin_p;
  const float vmax = *vmax_p;

  float v1[8], et[8];
  unsigned long long best = ~0ull;

#pragma unroll
  for (int j = 0; j < 8; ++j) {
    const int h = tid + j * 256;
    const size_t idx = (size_t)b * QIF_H + h;
    const float v = v_in[idx] + spike[idx];
    const float e = eta_arr[idx];
    v1[j] = v;
    et[j] = e;

    const float se = sqrtf(fabsf(e) + QIF_EPS);
    const float vs =
        v + QIF_EPS * qif_sign(v) + 0.5f * QIF_EPS * qif_sign(v + QIF_EPS);
    float dt;
    if (e == 0.0f) {
      dt = (v > 0.0f) ? (1.0f / vs) : QIF_TINF;
    } else if (e < 0.0f) {
      if (v > se) {
        float r = se / vs;
        r = fminf(fmaxf(r, -1.0f + QIF_EPS), 1.0f - QIF_EPS);
        dt = atanhf(r) / se;
      } else {
        dt = QIF_TINF;
      }
    } else {
      float a = atanf(se / vs);
      if (a < 0.0f) a += QIF_PI;  // python-style mod(a, pi), a in (-pi/2,pi/2)
      dt = a / se;
    }
    // dt >= 0 always -> float bits are order-preserving. Pack (bits, h):
    // min over u64 gives min dt with smallest index on ties (first argmin).
    const unsigned long long key =
        ((unsigned long long)__float_as_uint(dt) << 32) | (unsigned int)h;
    best = key < best ? key : best;
  }

  __shared__ unsigned long long sk[256];
  sk[tid] = best;
  __syncthreads();
#pragma unroll
  for (int s = 128; s > 0; s >>= 1) {
    if (tid < s) {
      const unsigned long long o = sk[tid + s];
      if (o < sk[tid]) sk[tid] = o;
    }
    __syncthreads();
  }
  const unsigned long long win = sk[0];
  const float min_dt = __uint_as_float((unsigned int)(win >> 32));
  const int k = (int)(unsigned int)win;
  const float dtt = min_dt + syn_delay;

#pragma unroll
  for (int j = 0; j < 8; ++j) {
    const float v = v1[j];
    const float e = et[j];
    const float se = sqrtf(fabsf(e) + QIF_EPS);
    float vn;
    if (e == 0.0f) {
      vn = v / (1.0f - dtt * v);
    } else if (e < 0.0f) {
      const float th = tanhf(se * dtt);
      vn = (v - se * th) / (1.0f - th * v / se);
    } else {
      const float ta = tanf(se * dtt);
      vn = (v + se * ta) / (1.0f - ta * v / se);
    }
    vn = fminf(fmaxf(vn, -1e25f), 1e25f);

    const int h = tid + j * 256;
    vn += w_h2h[(size_t)h * QIF_H + k];  // w_h2h.T[k, h]

    const size_t idx = (size_t)b * QIF_H + h;
    v_out[idx] = vn;
    vclip_bf[idx] = (__bf16)fminf(fmaxf(vn, vmin), vmax);
  }

  if (tid == 0) {
    t_out[b] = t_in[b] + dtt;
    mindt_out[b] = min_dt;
    ((int*)k_out_as_f32)[b] = k;
  }
}

// ---------------------------------------------------------------------------
// output_spike[b, o] = w_h2o[o, k[b]]
// ---------------------------------------------------------------------------
__global__ __launch_bounds__(256) void spike_gather_kernel(
    const float* __restrict__ w_h2o, const int* __restrict__ kk,
    float* __restrict__ out) {
  const int b = blockIdx.y;
  const int o = blockIdx.x * blockDim.x + threadIdx.x;
  out[(size_t)b * QIF_O + o] = w_h2o[(size_t)o * QIF_H + kk[b]];
}

// ---------------------------------------------------------------------------
extern "C" void kernel_launch(void* const* d_in, const int* in_sizes, int n_in,
                              void* d_out, int out_size, void* d_ws,
                              size_t ws_size, hipStream_t stream) {
  (void)in_sizes; (void)n_in; (void)out_size; (void)ws_size;
  const int B = QIF_B, H = QIF_H, I = QIF_I, O = QIF_O;

  const float* in_spike  = (const float*)d_in[0];
  const float* in_cur    = (const float*)d_in[1];
  const float* v_in      = (const float*)d_in[2];
  const float* t_in      = (const float*)d_in[3];
  const float* w_i2h     = (const float*)d_in[4];
  const float* b_i2h     = (const float*)d_in[5];
  const float* w_h2h     = (const float*)d_in[6];
  const float* b_h2h     = (const float*)d_in[7];
  const float* w_h2o     = (const float*)d_in[8];
  const float* b_h2o     = (const float*)d_in[9];
  const float* syn_delay = (const float*)d_in[10];
  const float* vminp     = (const float*)d_in[11];
  const float* vmaxp     = (const float*)d_in[12];

  // Output layout (return order, flat f32; k stored as int bits):
  float* out_spike = (float*)d_out;                    // [B,O]
  float* out_pot   = out_spike + (size_t)B * O;        // [B,O]
  float* out_v     = out_pot + (size_t)B * O;          // [B,H]
  float* out_t     = out_v + (size_t)B * H;            // [B]
  float* out_md    = out_t + B;                        // [B]
  float* out_k     = out_md + B;                       // [B] (int bits)

  // Workspace layout:
  float*  eta     = (float*)d_ws;                      // [B,H] f32   64 MB
  __bf16* cur_bf  = (__bf16*)(eta + (size_t)B * H);    // [B,I] bf16  16 MB
  __bf16* wi_bf   = cur_bf + (size_t)B * I;            // [H,I] bf16   4 MB
  __bf16* wo_bf   = wi_bf + (size_t)H * I;             // [O,H] bf16   2 MB
  __bf16* vclip   = wo_bf + (size_t)O * H;             // [B,H] bf16  32 MB

  // 1) Convert GEMM operands to bf16.
  f32_to_bf16_kernel<<<2048, 256, 0, stream>>>(in_cur, cur_bf, B * I);
  f32_to_bf16_kernel<<<2048, 256, 0, stream>>>(w_i2h, wi_bf, H * I);
  f32_to_bf16_kernel<<<1024, 256, 0, stream>>>(w_h2o, wo_bf, O * H);

  // 2) eta = current @ w_i2h^T + b_i2h + b_h2h
  gemm_bf16_wmma<<<dim3(H / 128, B / 128), 256, 0, stream>>>(
      cur_bf, wi_bf, b_i2h, b_h2h, eta, B, H, I);

  // 3) Fused neuron update (also produces clipped bf16 v for GEMM2).
  neuron_kernel<<<B, 256, 0, stream>>>(in_spike, v_in, eta, w_h2h, t_in,
                                       syn_delay, vminp, vmaxp, out_v, vclip,
                                       out_t, out_md, out_k);

  // 4) output_potential = clip(v) @ w_h2o^T + b_h2o ; output_spike gather.
  gemm_bf16_wmma<<<dim3(O / 128, B / 128), 256, 0, stream>>>(
      vclip, wo_bf, b_h2o, nullptr, out_pot, B, O, H);
  spike_gather_kernel<<<dim3(O / 256, B), 256, 0, stream>>>(
      w_h2o, (const int*)out_k, out_spike);
}
